// AttnReadout_63316407877633
// MI455X (gfx1250) — matrix operand order repaired and checked by reading, not compiled, not run on hardware
//
#include <hip/hip_runtime.h>
#include <math.h>

typedef _Float16 half_t;
typedef __attribute__((ext_vector_type(16))) half_t v16h;
typedef __attribute__((ext_vector_type(8)))  float  v8f;

#define WMMA_F32_F16(a, b, c) \
    __builtin_amdgcn_wmma_f32_16x16x32_f16(false, (a), false, (b), (short)0, (c), false, false)

// ---------------------------------------------------------------------------
// Kernel W: convert a row-major f32 weight matrix [NKC*32, NCT*16] into the
// CDNA5 16-bit B-matrix per-lane layout, split into (hi, lo) f16 halves.
// Layout: idx = ((kc*NCT + ct)*32 + lane)*16 + j
//   lanes 0-15 : N = lane,    halves j -> K = kc*32 + (j<8 ? j : 8+j)
//   lanes 16-31: N = lane-16, halves j -> K = kc*32 + 8 + (j<8 ? j : 8+j)
// ---------------------------------------------------------------------------
__global__ void convert_b_kernel(const float* __restrict__ W,
                                 half_t* __restrict__ hi, half_t* __restrict__ lo,
                                 int NKC, int NCT, int ncols) {
    int total = NKC * NCT * 32 * 16;
    for (int idx = blockIdx.x * blockDim.x + threadIdx.x; idx < total;
         idx += gridDim.x * blockDim.x) {
        int j    = idx & 15;
        int lane = (idx >> 4) & 31;
        int t    = idx >> 9;           // kc*NCT + ct
        int ct   = t % NCT;
        int kc   = t / NCT;
        int k    = kc * 32 + ((j < 8) ? j : 8 + j) + ((lane < 16) ? 0 : 8);
        int c    = ct * 16 + (lane & 15);
        float f  = W[(size_t)k * ncols + c];
        half_t h = (half_t)f;
        hi[idx]  = h;
        lo[idx]  = (half_t)(f - (float)h);
    }
}

// ---------------------------------------------------------------------------
// Kernel B: segment starts via binary search (batch is sorted ascending).
// starts[g] = lower_bound(batch, g); starts[B] = N. Handles empty segments.
// ---------------------------------------------------------------------------
__global__ void starts_kernel(const int* __restrict__ batch, int* __restrict__ starts,
                              int N, int B) {
    int g = blockIdx.x * blockDim.x + threadIdx.x;
    if (g > B) return;
    if (g == B) { starts[B] = N; return; }
    int lo = 0, hi = N;
    while (lo < hi) {
        int mid = (lo + hi) >> 1;
        if (batch[mid] < g) lo = mid + 1; else hi = mid;
    }
    starts[g] = lo;
}

// ---------------------------------------------------------------------------
// Kernel A: s = relu(x@W1 + b1)@W2 + b2   (per-node attention logit)
// One wave handles a 16-row M-tile. Split-f16 WMMA (hi*hi + lo*hi + hi*lo)
// gives near-fp32 accuracy. Layer 2 fused in VALU + 16-lane shuffle reduce.
// ---------------------------------------------------------------------------
__global__ __launch_bounds__(256) void score_kernel(
    const float* __restrict__ x, const half_t* __restrict__ Bhi,
    const half_t* __restrict__ Blo, const float* __restrict__ b1,
    const float* __restrict__ W2, const float* __restrict__ b2,
    float* __restrict__ s, int N) {
    const int H = 256, NCT = 8, NKC = 8;
    int wave = threadIdx.x >> 5;
    int lane = threadIdx.x & 31;
    int m0 = (blockIdx.x * 8 + wave) * 16;
    if (m0 >= N) return;

    int mrow  = m0 + (lane & 15);
    int khalf = (lane < 16) ? 0 : 8;
    const float* xrow = x + (size_t)mrow * H;

    v8f acc[NCT];
    #pragma unroll
    for (int ct = 0; ct < NCT; ++ct)
        #pragma unroll
        for (int v = 0; v < 8; ++v) acc[ct][v] = 0.0f;

    for (int kc = 0; kc < NKC; ++kc) {
        int kb = kc * 32;
        v16h ahi, alo;
        #pragma unroll
        for (int j = 0; j < 8; ++j) {
            float f0 = xrow[kb + khalf + j];
            float f1 = xrow[kb + 16 + khalf + j];
            half_t h0 = (half_t)f0, h1 = (half_t)f1;
            ahi[j]     = h0;
            ahi[8 + j] = h1;
            alo[j]     = (half_t)(f0 - (float)h0);
            alo[8 + j] = (half_t)(f1 - (float)h1);
        }
        #pragma unroll
        for (int ct = 0; ct < NCT; ++ct) {
            size_t boff = (((size_t)(kc * NCT + ct)) * 32 + lane) * 16;
            v16h bhi = *(const v16h*)(Bhi + boff);
            v16h blo = *(const v16h*)(Blo + boff);
            acc[ct] = WMMA_F32_F16(ahi, bhi, acc[ct]);
            acc[ct] = WMMA_F32_F16(alo, bhi, acc[ct]);
            acc[ct] = WMMA_F32_F16(ahi, blo, acc[ct]);
        }
    }

    // Layer 2: relu(h) @ W2, fused. acc[ct][v]: lanes<16 -> row v, lanes>=16 -> row v+8;
    // column = ct*16 + (lane&15).
    int lcol = lane & 15;
    float part[8];
    #pragma unroll
    for (int v = 0; v < 8; ++v) part[v] = 0.0f;
    #pragma unroll
    for (int ct = 0; ct < NCT; ++ct) {
        int col = ct * 16 + lcol;
        float bb = b1[col];
        float w2 = W2[col];
        #pragma unroll
        for (int v = 0; v < 8; ++v) {
            float h = fmaxf(acc[ct][v] + bb, 0.0f);
            part[v] += h * w2;
        }
    }
    float bias2 = b2[0];
    #pragma unroll
    for (int v = 0; v < 8; ++v) {
        float r = part[v];
        r += __shfl_xor(r, 1);
        r += __shfl_xor(r, 2);
        r += __shfl_xor(r, 4);
        r += __shfl_xor(r, 8);   // reduces within each 16-lane half
        if (lane == 0)  s[m0 + v]     = r + bias2;
        if (lane == 16) s[m0 + 8 + v] = r + bias2;
    }
}

// ---------------------------------------------------------------------------
// Kernel C: per-graph segment softmax + mean/attn/max/top-k pools.
// One block (256 threads == H) per graph. Stable top-k via O(n^2) ranking
// in LDS with the reference's exact tie-break (stable lexsort on (-w, idx)).
// ---------------------------------------------------------------------------
#define MAXSEG 2048
__global__ __launch_bounds__(256) void pool_kernel(
    const float* __restrict__ x, const float* __restrict__ s,
    float* __restrict__ w, const int* __restrict__ starts,
    float* __restrict__ pools) {
    const int H = 256;
    int g = blockIdx.x;
    int tid = threadIdx.x;
    int i0 = starts[g], i1 = starts[g + 1];
    int n = i1 - i0;
    float* prow = pools + (size_t)g * 4 * H;

    __shared__ float red[256];
    __shared__ float wbuf[MAXSEG];
    __shared__ unsigned char selb[MAXSEG];
    __shared__ float s_max, s_den;

    if (n <= 0) {   // empty segment: all pools are zero (uniform block exit)
        for (int h = tid; h < 4 * H; h += 256) prow[h] = 0.0f;
        return;
    }

    // --- segment max of s ---
    float m = -__builtin_inff();
    for (int i = tid; i < n; i += 256) m = fmaxf(m, s[i0 + i]);
    red[tid] = m; __syncthreads();
    for (int off = 128; off > 0; off >>= 1) {
        if (tid < off) red[tid] = fmaxf(red[tid], red[tid + off]);
        __syncthreads();
    }
    if (tid == 0) s_max = red[0];
    __syncthreads();
    m = s_max;

    // --- exp + denom, then normalize w (each thread owns i%256==tid) ---
    float d = 0.0f;
    for (int i = tid; i < n; i += 256) {
        float e = expf(s[i0 + i] - m);
        w[i0 + i] = e;
        d += e;
    }
    red[tid] = d; __syncthreads();
    for (int off = 128; off > 0; off >>= 1) {
        if (tid < off) red[tid] += red[tid + off];
        __syncthreads();
    }
    if (tid == 0) s_den = red[0];
    __syncthreads();
    float invd = 1.0f / s_den;
    for (int i = tid; i < n; i += 256) w[i0 + i] *= invd;

    // --- mean / attn / max pools: thread == feature, chunk w through LDS ---
    float accm = 0.0f, acca = 0.0f, accx = -__builtin_inff();
    for (int c0 = 0; c0 < n; c0 += 256) {
        int cn = min(256, n - c0);
        __syncthreads();
        if (tid < cn) wbuf[tid] = w[i0 + c0 + tid];
        __syncthreads();
        for (int j = 0; j < cn; ++j) {
            float xv = x[(size_t)(i0 + c0 + j) * H + tid];
            accm += xv;
            acca += wbuf[j] * xv;
            accx = fmaxf(accx, xv);
        }
    }
    prow[tid]         = accm / (float)n;
    prow[H + tid]     = acca;
    prow[2 * H + tid] = accx;

    // --- top-k: rank each node within segment (stable), select rank < k ---
    int nc = min(n, MAXSEG);
    __syncthreads();
    for (int i = tid; i < nc; i += 256) wbuf[i] = w[i0 + i];
    __syncthreads();
    int k = (int)ceilf(0.05f * (float)n);
    k = max(k, 5); k = min(k, 64); k = min(k, n);
    for (int a = tid; a < nc; a += 256) {
        float wa = wbuf[a];
        int r = 0;
        for (int b = 0; b < nc; ++b) {
            float wb = wbuf[b];
            r += (wb > wa) || (wb == wa && b < a);
        }
        selb[a] = (r < k) ? 1 : 0;
    }
    __syncthreads();
    float acct = 0.0f;
    for (int i = 0; i < nc; ++i)
        if (selb[i]) acct += x[(size_t)(i0 + i) * H + tid];
    prow[3 * H + tid] = acct / (float)max(k, 1);
}

// ---------------------------------------------------------------------------
// Kernel D: out = relu(pools[512,1024] @ Wf[1024,256] + bf), split-f16 WMMA.
// ---------------------------------------------------------------------------
__global__ __launch_bounds__(256) void final_kernel(
    const float* __restrict__ pools, const half_t* __restrict__ Bhi,
    const half_t* __restrict__ Blo, const float* __restrict__ bf,
    float* __restrict__ out, int M) {
    const int K = 1024, NOUT = 256, NKC = 32, NCT = 16;
    int wave = threadIdx.x >> 5;
    int lane = threadIdx.x & 31;
    int m0 = (blockIdx.x * 8 + wave) * 16;
    if (m0 >= M) return;

    int mrow  = m0 + (lane & 15);
    int khalf = (lane < 16) ? 0 : 8;
    int lcol  = lane & 15;
    const float* arow = pools + (size_t)mrow * K;

    for (int ct = 0; ct < NCT; ++ct) {
        v8f acc;
        #pragma unroll
        for (int v = 0; v < 8; ++v) acc[v] = 0.0f;
        for (int kc = 0; kc < NKC; ++kc) {
            int kb = kc * 32;
            v16h ahi, alo;
            #pragma unroll
            for (int j = 0; j < 8; ++j) {
                float f0 = arow[kb + khalf + j];
                float f1 = arow[kb + 16 + khalf + j];
                half_t h0 = (half_t)f0, h1 = (half_t)f1;
                ahi[j]     = h0;
                ahi[8 + j] = h1;
                alo[j]     = (half_t)(f0 - (float)h0);
                alo[8 + j] = (half_t)(f1 - (float)h1);
            }
            size_t boff = (((size_t)(kc * NCT + ct)) * 32 + lane) * 16;
            v16h bhi = *(const v16h*)(Bhi + boff);
            v16h blo = *(const v16h*)(Blo + boff);
            acc = WMMA_F32_F16(ahi, bhi, acc);
            acc = WMMA_F32_F16(alo, bhi, acc);
            acc = WMMA_F32_F16(ahi, blo, acc);
        }
        int colg = ct * 16 + lcol;
        float bias = bf[colg];
        #pragma unroll
        for (int v = 0; v < 8; ++v) {
            float r = fmaxf(acc[v] + bias, 0.0f);
            int row = (lane < 16) ? (m0 + v) : (m0 + 8 + v);
            out[(size_t)row * NOUT + colg] = r;
        }
    }
}

// ---------------------------------------------------------------------------
// Host launcher
// ---------------------------------------------------------------------------
static inline size_t align_up(size_t v, size_t a) { return (v + a - 1) & ~(a - 1); }

extern "C" void kernel_launch(void* const* d_in, const int* in_sizes, int n_in,
                              void* d_out, int out_size, void* d_ws, size_t ws_size,
                              hipStream_t stream) {
    const float* x     = (const float*)d_in[0];
    const int*   batch = (const int*)d_in[1];
    const float* W1    = (const float*)d_in[2];
    const float* b1    = (const float*)d_in[3];
    const float* W2    = (const float*)d_in[4];
    const float* b2    = (const float*)d_in[5];
    const float* Wf    = (const float*)d_in[6];
    const float* bf    = (const float*)d_in[7];
    (void)n_in; (void)ws_size;

    const int H = 2 * in_sizes[3];          // b1 has H/2 entries -> H = 256
    const int N = in_sizes[1];              // batch has N entries
    const int B = out_size / H;             // output is [B, H]

    // workspace carve-up
    char* ws = (char*)d_ws;
    size_t off = 0;
    float* s_buf = (float*)(ws + off); off = align_up(off + (size_t)N * 4, 256);
    float* w_buf = (float*)(ws + off); off = align_up(off + (size_t)N * 4, 256);
    int* starts  = (int*)(ws + off);   off = align_up(off + (size_t)(B + 1) * 4, 256);
    float* pools = (float*)(ws + off); off = align_up(off + (size_t)B * 4 * H * 4, 256);
    half_t* w1hi = (half_t*)(ws + off); off = align_up(off + (size_t)256 * 128 * 2, 256);
    half_t* w1lo = (half_t*)(ws + off); off = align_up(off + (size_t)256 * 128 * 2, 256);
    half_t* wfhi = (half_t*)(ws + off); off = align_up(off + (size_t)1024 * 256 * 2, 256);
    half_t* wflo = (half_t*)(ws + off); off = align_up(off + (size_t)1024 * 256 * 2, 256);

    // 1) weight pre-swizzle into WMMA B-layout (hi/lo f16)
    convert_b_kernel<<<128, 256, 0, stream>>>(W1, w1hi, w1lo, 8, 8, 128);
    convert_b_kernel<<<512, 256, 0, stream>>>(Wf, wfhi, wflo, 32, 16, 256);

    // 2) segment starts
    starts_kernel<<<(B + 1 + 255) / 256, 256, 0, stream>>>(batch, starts, N, B);

    // 3) node attention logits via split-f16 WMMA
    score_kernel<<<(N + 127) / 128, 256, 0, stream>>>(x, w1hi, w1lo, b1, W2, b2, s_buf, N);

    // 4) per-graph softmax + pools + stable top-k
    pool_kernel<<<B, 256, 0, stream>>>(x, s_buf, w_buf, starts, pools);

    // 5) final fused GEMM + bias + relu
    final_kernel<<<(B + 127) / 128, 256, 0, stream>>>(pools, wfhi, wflo, bf,
                                                      (float*)d_out, B);
}
